// RNN_42374147342656
// MI455X (gfx1250) — matrix operand order repaired and checked by reading, not compiled, not run on hardware
//
#include <hip/hip_runtime.h>
#include <hip/hip_bf16.h>
#include <math.h>

// ---------------------------------------------------------------------------
// Problem constants: T = B*S = 2048, H = 1024, D = 300, V_G = 40000, V_S = 25000
// ---------------------------------------------------------------------------
#define TT   2048
#define HH   1024
#define DD   300
#define VG   40000
#define VS   25000

typedef __attribute__((ext_vector_type(2)))  float   v2f;
typedef __attribute__((ext_vector_type(4)))  float   v4f;
typedef __attribute__((ext_vector_type(8)))  float   v8f;
typedef __attribute__((ext_vector_type(8)))  __bf16  v8bf;
typedef __attribute__((ext_vector_type(16))) __bf16  v16bf;

// ---------------------------------------------------------------------------
// Kernel 1: c_x = X[0,0:300] @ W[0:300, :]  -> written as hs row 0
// ---------------------------------------------------------------------------
__global__ void cx_kernel(const float* __restrict__ X,
                          const float* __restrict__ W,
                          float* __restrict__ hs0) {
    int h = blockIdx.x * blockDim.x + threadIdx.x;   // 0..1023
    float acc = 0.f;
    #pragma unroll 4
    for (int k = 0; k < DD; ++k)
        acc = fmaf(X[k], W[k * HH + h], acc);
    hs0[h] = acc;
}

// ---------------------------------------------------------------------------
// Kernel 2: fp32 WMMA GEMM, N = K = 1024 fixed:
//   C[M x 1024] = A[M x 1024] @ B[1024 x 1024] (+ broadcast row u if != null)
// A rows are loaded unconditionally (memory always valid; garbage rows only
// influence discarded C rows). Stores: uniform full-tile fast path (one
// scalar branch), masked path only for the n<16 doubling steps.
// Block = 128 threads = 4 waves, block tile 16x64.
// ---------------------------------------------------------------------------
__global__ void wmma_gemm_f32(const float* __restrict__ A,
                              const float* __restrict__ B,
                              float* __restrict__ C,
                              const float* __restrict__ u,
                              int M) {
    constexpr int N = HH, K = HH;
    const int lane = threadIdx.x & 31;
    const int wave = threadIdx.x >> 5;          // 0..3
    const int half = lane >> 4;                 // 0 or 1
    const int l    = lane & 15;
    const int n0   = blockIdx.x * 64 + wave * 16;
    const int m0   = blockIdx.y * 16;

    v8f acc = {};
    const float* __restrict__ Arow = A + (size_t)(m0 + l) * K;   // always valid memory
    const float* __restrict__ Bcol = B + n0 + l;

    #pragma unroll 4
    for (int k0 = 0; k0 < K; k0 += 4) {
        const int ka = k0 + half * 2;
        // A 16x4 fragment: lane l holds M=l; VGPR0/1 = K = 2*half, 2*half+1
        v2f a = *(const v2f*)(Arow + ka);
        // B 4x16 fragment: lane l holds N=n0+l
        v2f b;
        b.x = Bcol[ka * N];
        b.y = Bcol[(ka + 1) * N];
        acc = __builtin_amdgcn_wmma_f32_16x16x4_f32(false, a, false, b,
                                                    (short)0, acc, false, false);
    }

    const float uadd = (u != nullptr) ? u[n0 + l] : 0.f;
    if (m0 + 16 <= M) {                         // uniform: full tile, no masking
        #pragma unroll
        for (int i = 0; i < 8; ++i)
            C[(size_t)(m0 + half * 8 + i) * N + n0 + l] = acc[i] + uadd;
    } else {                                    // only for n < 16 doubling steps
        #pragma unroll
        for (int i = 0; i < 8; ++i) {
            const int m = m0 + half * 8 + i;
            if (m < M)
                C[(size_t)m * N + n0 + l] = acc[i] + uadd;
        }
    }
}

// ---------------------------------------------------------------------------
// Kernel 3: split hs (fp32) into bf16 hi/lo limb planes (hi + lo ~= fp32).
// ---------------------------------------------------------------------------
__global__ void split_bf16(const float* __restrict__ src,
                           __bf16* __restrict__ hi,
                           __bf16* __restrict__ lo, int n) {
    int i = blockIdx.x * blockDim.x + threadIdx.x;
    if (i < n) {
        float x = src[i];
        __bf16 h = (__bf16)x;
        hi[i] = h;
        lo[i] = (__bf16)(x - (float)h);
    }
}

// ---------------------------------------------------------------------------
// Kernel 4: big logits GEMM, split-bf16:
//   out[T x V] = hs[T x 1024] @ Wt[V x 1024]^T + bias[V]
// A pre-split (hsHi/hsLo bf16). Wt K-chunks (64 cols x 32 k) are converted
// ONCE per block into LDS limb planes. Register-staged pipeline: next chunk's
// global loads (A fragments + B fp32) issue right after the fill barrier and
// overlap the 12 wmmas of the current chunk. 3 limb products (hi*hi + hi*lo +
// lo*hi) accumulate in fp32 via v_wmma_f32_16x16x32_bf16. Block = 256 thr =
// 8 waves, block tile 128x64, wave strip 16x64 (A fragment reused 4x).
// Logits stores are non-temporal so the 532MB output stream doesn't evict the
// Wg/Ws working set (164MB, L2-resident across the 16 row-block passes).
// ---------------------------------------------------------------------------
__global__ void wmma_gemm_bf16split(const __bf16* __restrict__ Ahi,
                                    const __bf16* __restrict__ Alo,
                                    const float* __restrict__ Wt,
                                    const float* __restrict__ bias,
                                    float* __restrict__ out,
                                    int V) {
    constexpr int K = HH;                       // 1024
    constexpr int LDSTRIDE = 48;                // 32 + pad; 96B rows, 16B aligned
    __shared__ __align__(16) __bf16 Bhi[64][LDSTRIDE];
    __shared__ __align__(16) __bf16 Blo[64][LDSTRIDE];

    const int lane = threadIdx.x & 31;
    const int wave = threadIdx.x >> 5;          // 0..7
    const int half = lane >> 4;
    const int l    = lane & 15;
    const int m0   = blockIdx.y * 128 + wave * 16;   // row tile (< 2048 always)
    const int v0   = blockIdx.x * 64;                // col base

    // fill-phase role: each thread converts 8 consecutive k of one Wt row
    const int fv  = threadIdx.x >> 2;           // 0..63
    const int fkk = (threadIdx.x & 3) * 8;      // 0,8,16,24
    int vr = v0 + fv; if (vr > V - 1) vr = V - 1;          // clamp, branchless
    const float* __restrict__ Wrow = Wt + (size_t)vr * K + fkk;

    const __bf16* __restrict__ AhiRow = Ahi + (size_t)(m0 + l) * K;
    const __bf16* __restrict__ AloRow = Alo + (size_t)(m0 + l) * K;
    const int ha = half * 8;                    // A chunk offset
    const int hb = half * 16;                   // unused; B fragments also use ha

    v8f acc[4] = {{}, {}, {}, {}};

    // ---- preload chunk 0 into registers
    v4f  nt0  = *(const v4f*)(Wrow);
    v4f  nt1  = *(const v4f*)(Wrow + 4);
    v8bf nah0 = *(const v8bf*)(AhiRow + ha);
    v8bf nah1 = *(const v8bf*)(AhiRow + 16 + ha);
    v8bf nal0 = *(const v8bf*)(AloRow + ha);
    v8bf nal1 = *(const v8bf*)(AloRow + 16 + ha);

    for (int k0 = 0; k0 < K; k0 += 32) {
        const v4f  t0 = nt0,  t1 = nt1;
        const v8bf ah0 = nah0, ah1 = nah1, al0 = nal0, al1 = nal1;

        // convert current B chunk and store limb planes to LDS
        v8bf hv, lv;
        #pragma unroll
        for (int i = 0; i < 4; ++i) {
            __bf16 h0 = (__bf16)t0[i];
            hv[i]     = h0;
            lv[i]     = (__bf16)(t0[i] - (float)h0);
            __bf16 h1 = (__bf16)t1[i];
            hv[4 + i] = h1;
            lv[4 + i] = (__bf16)(t1[i] - (float)h1);
        }
        *(v8bf*)&Bhi[fv][fkk] = hv;
        *(v8bf*)&Blo[fv][fkk] = lv;
        __syncthreads();

        // issue NEXT chunk's global loads: they retire under the wmmas below
        if (k0 + 32 < K) {
            nt0  = *(const v4f*)(Wrow + k0 + 32);
            nt1  = *(const v4f*)(Wrow + k0 + 36);
            nah0 = *(const v8bf*)(AhiRow + k0 + 32 + ha);
            nah1 = *(const v8bf*)(AhiRow + k0 + 48 + ha);
            nal0 = *(const v8bf*)(AloRow + k0 + 32 + ha);
            nal1 = *(const v8bf*)(AloRow + k0 + 48 + ha);
        }

        // A 16x32 bf16 fragments (hi/lo):
        // lane<16: K = {k0..+7} U {k0+16..+23}; lane>=16: +8 on both chunks.
        v16bf ahi, alo;
        #pragma unroll
        for (int i = 0; i < 8; ++i) {
            ahi[i] = ah0[i]; ahi[8 + i] = ah1[i];
            alo[i] = al0[i]; alo[8 + i] = al1[i];
        }

        #pragma unroll
        for (int t = 0; t < 4; ++t) {
            // B 32x16 fragment: lane supplies column N = t*16+l,
            // K rows = {ha..ha+7} U {16+ha..16+ha+7} (same order as A).
            const __bf16* bp = &Bhi[t * 16 + l][ha];
            const __bf16* lp = &Blo[t * 16 + l][ha];
            v8bf bh0 = *(const v8bf*)bp;
            v8bf bh1 = *(const v8bf*)(bp + 16);
            v8bf bl0 = *(const v8bf*)lp;
            v8bf bl1 = *(const v8bf*)(lp + 16);
            v16bf bhi, blo;
            #pragma unroll
            for (int i = 0; i < 8; ++i) {
                bhi[i] = bh0[i]; bhi[8 + i] = bh1[i];
                blo[i] = bl0[i]; blo[8 + i] = bl1[i];
            }
            acc[t] = __builtin_amdgcn_wmma_f32_16x16x32_bf16(false, ahi, false, bhi,
                                                             (short)0, acc[t], false, false);
            acc[t] = __builtin_amdgcn_wmma_f32_16x16x32_bf16(false, ahi, false, blo,
                                                             (short)0, acc[t], false, false);
            acc[t] = __builtin_amdgcn_wmma_f32_16x16x32_bf16(false, alo, false, bhi,
                                                             (short)0, acc[t], false, false);
        }
        __syncthreads();                        // protect LDS before next fill
    }

    #pragma unroll
    for (int t = 0; t < 4; ++t) {
        const int v = v0 + t * 16 + l;
        if (v < V) {
            const float b = bias[v];
            #pragma unroll
            for (int i = 0; i < 8; ++i) {
                const int m = m0 + half * 8 + i;
                __builtin_nontemporal_store(acc[t][i] + b,
                                            out + (size_t)m * V + v);
            }
        }
    }
}

// ---------------------------------------------------------------------------
// Kernel 5: single-pass streaming logsumexp per row (float4, NT loads).
// Rows 0..2047 -> globals (width VG); rows 2048..4095 -> senses (VS).
// ---------------------------------------------------------------------------
__global__ void row_logsumexp(const float* __restrict__ out,
                              float* __restrict__ stat) {
    const int r = blockIdx.x;
    const float* row;
    int width;
    if (r < TT) { row = out + (size_t)r * VG;                          width = VG; }
    else        { row = out + (size_t)TT * VG + (size_t)(r - TT) * VS; width = VS; }

    const v4f* row4 = (const v4f*)row;
    const int  w4   = width >> 2;               // VG,VS both % 4 == 0
    float m = -INFINITY, s = 0.f;
    for (int c = threadIdx.x; c < w4; c += blockDim.x) {
        const v4f x4 = __builtin_nontemporal_load(row4 + c);
        #pragma unroll
        for (int j = 0; j < 4; ++j) {
            const float x = x4[j];
            if (x > m) { s = s * __expf(m - x) + 1.f; m = x; }
            else       { s += __expf(x - m); }
        }
    }
    __shared__ float sm[256], ss[256];
    const int tid = threadIdx.x;
    sm[tid] = m; ss[tid] = s;
    __syncthreads();
    for (int off = 128; off > 0; off >>= 1) {
        if (tid < off) {
            const float m2 = sm[tid + off], s2 = ss[tid + off];
            const float mm = fmaxf(sm[tid], m2);
            ss[tid] = ss[tid] * __expf(sm[tid] - mm) + s2 * __expf(m2 - mm);
            sm[tid] = mm;
        }
        __syncthreads();
    }
    if (tid == 0) stat[r] = sm[0] + __logf(ss[0]);
}

// ---------------------------------------------------------------------------
// Kernel 6: out[r, :] -= stat[r]   (finishes log_softmax), float4 + NT.
// ---------------------------------------------------------------------------
__global__ void sub_stat(float* __restrict__ out,
                         const float* __restrict__ stat,
                         int width) {
    const int r  = blockIdx.y;
    const int c4 = blockIdx.x * blockDim.x + threadIdx.x;
    const int w4 = width >> 2;
    if (c4 < w4) {
        v4f* row4 = (v4f*)(out + (size_t)r * width);
        const float s = stat[r];
        v4f x = __builtin_nontemporal_load(row4 + c4);
        x[0] -= s; x[1] -= s; x[2] -= s; x[3] -= s;
        __builtin_nontemporal_store(x, row4 + c4);
    }
}

// ---------------------------------------------------------------------------
// Host orchestration.
// ws (floats): hs[2048*1024] | powA[1024^2] | powB[1024^2] | stat[4096]
//              | hsHi, hsLo (bf16, 2048*1024 each)          (~24.3 MB)
// Scan via log-depth doubling:  hs[n:2n] = hs[0:n] @ W^n + bcast(hs[n-1]),
// with squarings W^(2n) = W^n @ W^n interleaved (fp32 WMMA for accuracy).
// ---------------------------------------------------------------------------
extern "C" void kernel_launch(void* const* d_in, const int* in_sizes, int n_in,
                              void* d_out, int out_size, void* d_ws, size_t ws_size,
                              hipStream_t stream) {
    (void)in_sizes; (void)n_in; (void)out_size; (void)ws_size;
    const float* X  = (const float*)d_in[1];
    const float* W  = (const float*)d_in[2];
    const float* Wg = (const float*)d_in[3];
    const float* bg = (const float*)d_in[4];
    const float* Ws = (const float*)d_in[5];
    const float* bs = (const float*)d_in[6];
    float* out = (float*)d_out;

    float*  ws   = (float*)d_ws;
    float*  hs   = ws;                                  // 2048*1024 f32
    float*  powA = hs + (size_t)TT * HH;                // 1024*1024 f32
    float*  powB = powA + (size_t)HH * HH;              // 1024*1024 f32
    float*  stat = powB + (size_t)HH * HH;              // 4096 f32
    __bf16* hsHi = (__bf16*)(stat + 4096);              // 2048*1024 bf16
    __bf16* hsLo = hsHi + (size_t)TT * HH;              // 2048*1024 bf16

    const float* Whh = W + (size_t)DD * HH;             // W[300:,:]

    // 1) hs[0] = c_x
    cx_kernel<<<HH / 256, 256, 0, stream>>>(X, W, hs);

    // 2) log-depth doubling of the scan (fp32 WMMA)
    const float* cur = Whh;
    float* bufs[2] = {powA, powB};
    for (int s = 0; s <= 10; ++s) {
        const int n = 1 << s;
        dim3 gd(HH / 64, (n + 15) / 16);
        wmma_gemm_f32<<<gd, 128, 0, stream>>>(hs, cur,
                                              hs + (size_t)n * HH,
                                              hs + (size_t)(n - 1) * HH, n);
        if (s < 10) {
            float* nxt = bufs[s & 1];
            dim3 gs(HH / 64, HH / 16);
            wmma_gemm_f32<<<gs, 128, 0, stream>>>(cur, cur, nxt, nullptr, HH);
            cur = nxt;
        }
    }

    // 3) split hs into bf16 limb planes
    {
        const int n = TT * HH;
        split_bf16<<<(n + 255) / 256, 256, 0, stream>>>(hs, hsHi, hsLo, n);
    }

    // 4) logits GEMMs (split-bf16 WMMA, LDS-staged B limbs), bias fused
    {
        dim3 g((VG + 63) / 64, TT / 128);
        wmma_gemm_bf16split<<<g, 256, 0, stream>>>(hsHi, hsLo, Wg, bg, out, VG);
    }
    float* outS = out + (size_t)TT * VG;
    {
        dim3 g((VS + 63) / 64, TT / 128);
        wmma_gemm_bf16split<<<g, 256, 0, stream>>>(hsHi, hsLo, Ws, bs, outS, VS);
    }

    // 5) log-softmax: streaming logsumexp then subtract
    row_logsumexp<<<2 * TT, 256, 0, stream>>>(out, stat);
    {
        dim3 g((VG / 4 + 255) / 256, TT);
        sub_stat<<<g, 256, 0, stream>>>(out, stat, VG);
    }
    {
        dim3 g((VS / 4 + 255) / 256, TT);
        sub_stat<<<g, 256, 0, stream>>>(outS, stat + TT, VS);
    }
}
